// LocalAttentionWrapper_19275813224539
// MI455X (gfx1250) — matrix operand order repaired and checked by reading, not compiled, not run on hardware
//
#include <hip/hip_runtime.h>

typedef __attribute__((ext_vector_type(16))) __bf16 v16bf;
typedef __attribute__((ext_vector_type(8)))  __bf16 v8bf;
typedef __attribute__((ext_vector_type(8)))  float  v8f;
typedef __attribute__((ext_vector_type(4)))  int    v4i;

// ---------------------------------------------------------------------------
// Problem constants (from reference): B=4, S=4096, D=1024, H=16, dh=64,
// CHUNK=64, LOOKBACK=64. Tokens M = 16384.
// ---------------------------------------------------------------------------
#define TOKENS   16384
#define DMODEL   1024
#define QKV_N    3072
#define NHEAD    16
#define DHEAD    64
#define NCHUNK   64
#define SEQ      4096

// ---------------------------------------------------------------------------
// gfx1250 async global->LDS copy (ASYNCcnt) with safe fallback.
// ROCm clang signature: (AS1 int4* gaddr, AS3 int4* laddr, imm offset, imm cpol)
// ---------------------------------------------------------------------------
#if defined(__HIP_DEVICE_COMPILE__) &&                                   \
    __has_builtin(__builtin_amdgcn_global_load_async_to_lds_b128) &&     \
    __has_builtin(__builtin_amdgcn_s_wait_asynccnt)
#define USE_ASYNC_LDS 1
#else
#define USE_ASYNC_LDS 0
#endif

__device__ __forceinline__ void cp16_g2l(__bf16* l, const __bf16* g) {
#if USE_ASYNC_LDS
  __builtin_amdgcn_global_load_async_to_lds_b128(
      (__attribute__((address_space(1))) v4i*)(g),
      (__attribute__((address_space(3))) v4i*)(l), 0, 0);
#else
  *(v8bf*)l = *(const v8bf*)g;
#endif
}

__device__ __forceinline__ void wait_async_copies() {
#if USE_ASYNC_LDS
  __builtin_amdgcn_s_wait_asynccnt(0);
#endif
}

// ---------------------------------------------------------------------------
// Kernel 1: f32 [K,N] -> bf16 [N,K] transpose (weights; tiny, one-shot)
// ---------------------------------------------------------------------------
__global__ void transpose_f32_to_bf16(const float* __restrict__ src,
                                      __bf16* __restrict__ dst,
                                      int K, int N) {
  size_t i = (size_t)blockIdx.x * 256 + threadIdx.x;
  if (i < (size_t)K * N) {
    int k = (int)(i / N);
    int n = (int)(i % N);
    dst[(size_t)n * K + k] = (__bf16)src[i];
  }
}

// ---------------------------------------------------------------------------
// Kernel 2: fused residual add + RMSNorm.
// res_out (f32, second reference output), x_out (bf16 GEMM operand).
// One block (256 thr = 8 waves) per token row of 1024.
// ---------------------------------------------------------------------------
__global__ void fused_add_rmsnorm_kernel(const float* __restrict__ h,
                                         const float* __restrict__ r,
                                         const float* __restrict__ w,
                                         float* __restrict__ res_out,
                                         __bf16* __restrict__ x_out) {
  const int row = blockIdx.x;
  const int tid = threadIdx.x;            // 256
  const size_t base = (size_t)row * DMODEL;
  float v[4];
  float ss = 0.f;
#pragma unroll
  for (int i = 0; i < 4; ++i) {
    int idx = tid + i * 256;
    float a = h[base + idx] + r[base + idx];
    v[i] = a;
    res_out[base + idx] = a;
    ss += a * a;
  }
  // wave32 reduction
#pragma unroll
  for (int m = 16; m >= 1; m >>= 1) ss += __shfl_xor(ss, m, 32);
  __shared__ float red[8];
  __shared__ float s_inv;
  if ((tid & 31) == 0) red[tid >> 5] = ss;
  __syncthreads();
  if (tid == 0) {
    float t = 0.f;
#pragma unroll
    for (int i = 0; i < 8; ++i) t += red[i];
    s_inv = rsqrtf(t * (1.0f / (float)DMODEL) + 1e-5f);
  }
  __syncthreads();
  const float inv = s_inv;
#pragma unroll
  for (int i = 0; i < 4; ++i) {
    int idx = tid + i * 256;
    x_out[base + idx] = (__bf16)(v[i] * inv * w[idx]);
  }
}

// ---------------------------------------------------------------------------
// Kernel 3/5: bf16 GEMM, C = A[M,K] * Bt[N,K]^T, f32 accumulate via WMMA.
// Block: 256 thr = 8 waves (2 M x 4 N); block tile 128 x 128; K step 32.
// A/B tiles double-buffered in LDS via async global->LDS copies.
// Per wave: 64(M) x 32(N) output = 8 f32 accumulators; 8 WMMA per k-step
// fed by 6 LDS fragment loads.
// Fragment addressing per ISA 16-bit layouts:
//   A: lane<16 -> K 0-7 / 16-23 ; lane>=16 -> +8. Row = M + (lane&15).
//   B: lane column = N + (lane&15); same K order as A (K 0-7 / 16-23, +8 for hl).
// ---------------------------------------------------------------------------
template <bool OUT_F32>
__global__ void gemm_bf16_wmma(const __bf16* __restrict__ A,
                               const __bf16* __restrict__ Bt,
                               void* __restrict__ Cv,
                               int M, int N, int K) {
  __shared__ __attribute__((aligned(16))) __bf16 aBuf[2][128 * 32];
  __shared__ __attribute__((aligned(16))) __bf16 bBuf[2][128 * 32];

  const int tid   = threadIdx.x;
  const int lane  = tid & 31;
  const int wave  = tid >> 5;
  const int waveM = wave >> 2;            // 0..1
  const int waveN = wave & 3;             // 0..3
  const int hl    = lane >> 4;
  const int l15   = lane & 15;
  const int mblk  = blockIdx.x * 128;
  const int nblk  = blockIdx.y * 128;

  // stage one 128x32 A tile + 128x32 B tile (16B chunks, 4 per thread)
  auto stage = [&](int buf, int k0) {
#pragma unroll
    for (int u = 0; u < 2; ++u) {
      int c   = tid * 2 + u;        // 0..511
      int r   = c >> 2;             // 0..127
      int off = (c & 3) * 8;        // 0,8,16,24
      cp16_g2l(&aBuf[buf][r * 32 + off], A  + (size_t)(mblk + r) * K + k0 + off);
      cp16_g2l(&bBuf[buf][r * 32 + off], Bt + (size_t)(nblk + r) * K + k0 + off);
    }
  };

  const int KT = K >> 5;            // k-steps of 32
  v8f acc[4][2] = {};

  stage(0, 0);
  for (int kt = 0; kt < KT; ++kt) {
    wait_async_copies();
    __syncthreads();                 // tile (kt&1) visible to all waves
    if (kt + 1 < KT) stage((kt + 1) & 1, (kt + 1) << 5);

    const __bf16* aT = aBuf[kt & 1];
    const __bf16* bT = bBuf[kt & 1];

    v16bf af[4];
#pragma unroll
    for (int fi = 0; fi < 4; ++fi) {
      const __bf16* ap = &aT[(waveM * 64 + fi * 16 + l15) * 32 + hl * 8];
      ((v8bf*)&af[fi])[0] = *(const v8bf*)(ap);
      ((v8bf*)&af[fi])[1] = *(const v8bf*)(ap + 16);
    }
#pragma unroll
    for (int j = 0; j < 2; ++j) {
      const __bf16* bp = &bT[(waveN * 32 + j * 16 + l15) * 32 + hl * 8];
      v16bf bf;
      ((v8bf*)&bf)[0] = *(const v8bf*)(bp);
      ((v8bf*)&bf)[1] = *(const v8bf*)(bp + 16);
#pragma unroll
      for (int fi = 0; fi < 4; ++fi)
        acc[fi][j] = __builtin_amdgcn_wmma_f32_16x16x32_bf16(
            false, af[fi], false, bf, (short)0, acc[fi][j], false, false);
    }
    __syncthreads();                 // done reading tile before it is reused
  }

#pragma unroll
  for (int fi = 0; fi < 4; ++fi) {
#pragma unroll
    for (int j = 0; j < 2; ++j) {
#pragma unroll
      for (int r = 0; r < 8; ++r) {
        int row = mblk + waveM * 64 + fi * 16 + r + hl * 8;
        int col = nblk + waveN * 32 + j * 16 + l15;
        if constexpr (OUT_F32)
          ((float*)Cv)[(size_t)row * N + col] = acc[fi][j][r];
        else
          ((__bf16*)Cv)[(size_t)row * N + col] = (__bf16)acc[fi][j][r];
      }
    }
  }
}

// ---------------------------------------------------------------------------
// Kernel 4: sliding-window attention for one (b, head, chunk).
// Block: 128 thr = 4 waves; each wave owns a 16-query strip.
// Dynamic LDS layout (bytes):
//   qS   [64][64]  bf16   @ 0      (8192)
//   kS   [128][64] bf16   @ 8192   (16384)
//   vtS  [64][136] bf16   @ 24576  (17408)   (v transposed, padded)
//   sS   [64][132] f32    @ 41984  (33792)   (scores, padded)
//   aS   [64][128] bf16   @ 75776  (16384)   (softmaxed attn)
// total 92160 bytes
// ---------------------------------------------------------------------------
#define ATTN_LDS_BYTES 92160

__global__ void local_attn_kernel(const __bf16* __restrict__ qkv,
                                  __bf16* __restrict__ out) {
  extern __shared__ char smem[];
  __bf16* qS  = (__bf16*)(smem);
  __bf16* kS  = (__bf16*)(smem + 8192);
  __bf16* vtS = (__bf16*)(smem + 24576);
  float*  sS  = (float*)(smem + 41984);
  __bf16* aS  = (__bf16*)(smem + 75776);

  int idx = blockIdx.x;
  const int c = idx % NCHUNK; idx /= NCHUNK;
  const int h = idx % NHEAD;
  const int b = idx / NHEAD;
  const int tid  = threadIdx.x;       // 128
  const int lane = tid & 31;
  const int wave = tid >> 5;
  const size_t tokbase = (size_t)b * SEQ + (size_t)c * 64;

  // --- stage q ---
  for (int i = tid; i < 64 * 64; i += 128) {
    int r = i >> 6, d = i & 63;
    qS[i] = qkv[(tokbase + r) * QKV_N + h * 64 + d];
  }
  // --- stage k2 (prev chunk + this chunk) and v2 transposed ---
  for (int i = tid; i < 128 * 64; i += 128) {
    int r = i >> 6, d = i & 63;     // r: key row 0..127 (r<64 => prev chunk)
    __bf16 kv, vv;
    if (c == 0 && r < 64) {
      kv = (__bf16)0.0f;
      vv = (__bf16)0.0f;
    } else {
      size_t tok = tokbase - 64 + r;
      kv = qkv[tok * QKV_N + 1024 + h * 64 + d];
      vv = qkv[tok * QKV_N + 2048 + h * 64 + d];
    }
    kS[i] = kv;
    vtS[d * 136 + r] = vv;
  }
  __syncthreads();

  const int hl  = lane >> 4;
  const int l15 = lane & 15;
  const int mrow = wave * 16 + l15;   // A-fragment row for this lane

  // --- phase 1: scores[64,128] = q @ k2^T  (scaled) ---
  {
    v8f acc[8] = {};
#pragma unroll
    for (int k0 = 0; k0 < DHEAD; k0 += 32) {
      v16bf a;
      const __bf16* ap = &qS[mrow * 64 + k0 + hl * 8];
      ((v8bf*)&a)[0] = *(const v8bf*)(ap);
      ((v8bf*)&a)[1] = *(const v8bf*)(ap + 16);
#pragma unroll
      for (int j = 0; j < 8; ++j) {
        const __bf16* bp = &kS[(j * 16 + l15) * 64 + k0 + hl * 8];
        v16bf bb;
        ((v8bf*)&bb)[0] = *(const v8bf*)(bp);
        ((v8bf*)&bb)[1] = *(const v8bf*)(bp + 16);
        acc[j] = __builtin_amdgcn_wmma_f32_16x16x32_bf16(
            false, a, false, bb, (short)0, acc[j], false, false);
      }
    }
#pragma unroll
    for (int j = 0; j < 8; ++j)
#pragma unroll
      for (int r = 0; r < 8; ++r)
        sS[(wave * 16 + r + hl * 8) * 132 + j * 16 + l15] = acc[j][r] * 0.125f;
  }
  __syncthreads();

  // --- softmax with sliding-window mask; thread t < 64 handles row t ---
  if (tid < 64) {
    const int i = tid;
    float mx = -1e30f;
    for (int j = 0; j < 128; ++j) {
      bool valid = (j >= i) && (j <= i + 64) && ((c > 0) || (j >= 64));
      float s = valid ? sS[i * 132 + j] : -1e30f;
      sS[i * 132 + j] = s;
      mx = fmaxf(mx, s);
    }
    float sum = 0.f;
    for (int j = 0; j < 128; ++j) {
      float e = __expf(sS[i * 132 + j] - mx);
      sS[i * 132 + j] = e;
      sum += e;
    }
    float inv = 1.0f / sum;
    for (int j = 0; j < 128; ++j)
      aS[i * 128 + j] = (__bf16)(sS[i * 132 + j] * inv);
  }
  __syncthreads();

  // --- phase 2: out[64,64] = attn @ v2 ---
  {
    v8f acc[4] = {};
#pragma unroll
    for (int k0 = 0; k0 < 128; k0 += 32) {
      v16bf a;
      const __bf16* ap = &aS[mrow * 128 + k0 + hl * 8];
      ((v8bf*)&a)[0] = *(const v8bf*)(ap);
      ((v8bf*)&a)[1] = *(const v8bf*)(ap + 16);
#pragma unroll
      for (int j = 0; j < 4; ++j) {
        const __bf16* bp = &vtS[(j * 16 + l15) * 136 + k0 + hl * 8];
        v16bf bb;
        ((v8bf*)&bb)[0] = *(const v8bf*)(bp);
        ((v8bf*)&bb)[1] = *(const v8bf*)(bp + 16);
        acc[j] = __builtin_amdgcn_wmma_f32_16x16x32_bf16(
            false, a, false, bb, (short)0, acc[j], false, false);
      }
    }
#pragma unroll
    for (int j = 0; j < 4; ++j)
#pragma unroll
      for (int r = 0; r < 8; ++r) {
        size_t tok = tokbase + wave * 16 + r + hl * 8;
        out[tok * DMODEL + h * 64 + j * 16 + l15] = (__bf16)acc[j][r];
      }
  }
}

// ---------------------------------------------------------------------------
// Host entry
// ---------------------------------------------------------------------------
extern "C" void kernel_launch(void* const* d_in, const int* in_sizes, int n_in,
                              void* d_out, int out_size, void* d_ws, size_t ws_size,
                              hipStream_t stream) {
  const float* hs    = (const float*)d_in[0];   // hidden_states [4,4096,1024]
  const float* resid = (const float*)d_in[1];   // residual
  const float* nw    = (const float*)d_in[2];   // norm_weight [1024]
  const float* wqkv  = (const float*)d_in[3];   // [1024, 3072]
  const float* wout  = (const float*)d_in[4];   // [1024, 1024]

  float* out     = (float*)d_out;                           // output #1
  float* res_out = out + (size_t)TOKENS * DMODEL;           // output #2

  char* ws = (char*)d_ws;
  __bf16* xb    = (__bf16*)ws;  ws += (size_t)TOKENS * DMODEL * 2;   // 33.6 MB
  __bf16* wqkvT = (__bf16*)ws;  ws += (size_t)QKV_N * DMODEL * 2;    //  6.3 MB
  __bf16* woutT = (__bf16*)ws;  ws += (size_t)DMODEL * DMODEL * 2;   //  2.1 MB
  __bf16* qkvb  = (__bf16*)ws;  ws += (size_t)TOKENS * QKV_N * 2;    // 100.7 MB
  __bf16* attb  = (__bf16*)ws;                                       // 33.6 MB

  // 1) weights -> bf16 transposed
  transpose_f32_to_bf16<<<(DMODEL * QKV_N + 255) / 256, 256, 0, stream>>>(
      wqkv, wqkvT, DMODEL, QKV_N);
  transpose_f32_to_bf16<<<(DMODEL * DMODEL + 255) / 256, 256, 0, stream>>>(
      wout, woutT, DMODEL, DMODEL);

  // 2) fused add + RMSNorm
  fused_add_rmsnorm_kernel<<<TOKENS, 256, 0, stream>>>(hs, resid, nw, res_out, xb);

  // 3) QKV projection: [16384,1024] x [1024,3072] -> bf16
  gemm_bf16_wmma<false><<<dim3(TOKENS / 128, QKV_N / 128), 256, 0, stream>>>(
      xb, wqkvT, (void*)qkvb, TOKENS, QKV_N, DMODEL);

  // 4) sliding-window attention: one block per (b, head, chunk)
  local_attn_kernel<<<4 * NHEAD * NCHUNK, 128, ATTN_LDS_BYTES, stream>>>(qkvb, attb);

  // 5) output projection: [16384,1024] x [1024,1024] -> f32 out
  gemm_bf16_wmma<true><<<dim3(TOKENS / 128, DMODEL / 128), 256, 0, stream>>>(
      attb, woutT, (void*)out, TOKENS, DMODEL, DMODEL);
}
